// ConditionalSmilesRnn_84327387889670
// MI455X (gfx1250) — compile-verified, matrix operand order checked by reading, not run on hardware
//
#include <hip/hip_runtime.h>
#include <hip/hip_bf16.h>
#include <stdint.h>

// Problem constants (from the reference)
#define B_   128
#define T_   64
#define H_   512
#define O_   47
#define P_   4
#define G_   2048      // 4*H gates
#define K0P  544       // layer-0 K padded: 512 (h) + 4 (props) + 28 zeros
#define OPAD 48        // decoder N padded to 3 tiles of 16
#define KE   64        // encoder K padded: 47 -> 64

#define NBLK 32        // 32 blocks * 8 waves = 256 waves = 8 Mtiles x 32 Jtiles
#define TPB_ 256

typedef __attribute__((ext_vector_type(16))) __bf16 bf16x16;
typedef __attribute__((ext_vector_type(8)))  float  f32x8;

// Opaque 64-bit value (redefined every t-iteration): adding it to a pointer
// defeats LICM/GVN across the recurrence loop WITHOUT destroying address-space
// inference (base stays a kernel arg -> loads stay global_load, not flat_load).
#define OPQ(x) asm volatile("" : "+s"(x))

// ---- workspace layout (bytes, all 256-aligned) ----
#define WS_WIH0  0ull            // [2048][544] bf16  = 2,228,224
#define WS_WHH0  2228224ull      // [2048][512] bf16  = 2,097,152
#define WS_WIHR  4325376ull      // [2][2048][512] bf16 = 4,194,304
#define WS_WHHR  8519680ull      // [2][2048][512] bf16 = 4,194,304
#define WS_DECP  12713984ull     // [48][512] bf16    = 49,152
#define WS_ENCP  12763136ull     // [512][64] bf16    = 65,536
#define WS_BIAS  12828672ull     // [3][2048] f32     = 24,576
#define WS_XCAT  12853248ull     // [128][544] bf16   = 139,264
#define WS_HBUF  12992512ull     // [3][2][128][512] bf16 = 786,432
#define WS_CBUF  13778944ull     // [3][128][512] f32 = 786,432
#define WS_LOGB  14565376ull     // [128][64] bf16    = 16,384
#define WS_BAR   14581760ull     // barrier state

__device__ __forceinline__ unsigned short f32_to_bf16u(float f) {
  unsigned u = __float_as_uint(f);
  unsigned r = 0x7FFFu + ((u >> 16) & 1u);   // round-to-nearest-even
  return (unsigned short)((u + r) >> 16);
}
__device__ __forceinline__ float sigm(float x) { return 1.0f / (1.0f + __expf(-x)); }

// A/B fragment for v_wmma_f32_16x16x32_bf16.
// Lane L holds row/col (L&15); elements 0..7 are K = k0..k0+7, elements 8..15
// are K = k0+16..k0+23, where k0 = 32*kIter + 8*(L>=16). Both halves are one
// 16B contiguous load since the source is [N,K] row-major -> two b128 loads.
__device__ __forceinline__ bf16x16 load_frag(const unsigned short* base, int ld,
                                             int r, int k0) {
  bf16x16 f;
  const uint4* p = (const uint4*)(base + (size_t)r * ld + k0);
  ((uint4*)&f)[0] = p[0];   // K: k0 .. k0+7
  ((uint4*)&f)[1] = p[2];   // K: k0+16 .. k0+23
  return f;
}

__device__ __forceinline__ f32x8 wmma_bf16(bf16x16 a, bf16x16 b, f32x8 c) {
  return __builtin_amdgcn_wmma_f32_16x16x32_bf16(false, a, false, b,
                                                 (short)0, c, false, false);
}

// Cooperative copy of a contiguous A panel (16 rows x ld bf16) into LDS.
__device__ __forceinline__ void stage16(unsigned short* dst_sh,
                                        const unsigned short* src, int elems) {
  const uint4* s = (const uint4*)src;
  uint4* d = (uint4*)dst_sh;
  const int n16 = elems >> 3;               // 8 bf16 per 16B chunk
  for (int i = threadIdx.x; i < n16; i += TPB_) d[i] = s[i];
}

// Grid-wide sense/generation barrier (state reset by prep each launch).
__device__ __forceinline__ void grid_sync(unsigned* bar, unsigned nblocks) {
  __threadfence();
  __syncthreads();
  if (threadIdx.x == 0) {
    unsigned g = __hip_atomic_load(&bar[1], __ATOMIC_RELAXED, __HIP_MEMORY_SCOPE_AGENT);
    unsigned a = __hip_atomic_fetch_add(&bar[0], 1u, __ATOMIC_ACQ_REL, __HIP_MEMORY_SCOPE_AGENT);
    if (a == nblocks - 1u) {
      __hip_atomic_store(&bar[0], 0u, __ATOMIC_RELAXED, __HIP_MEMORY_SCOPE_AGENT);
      __hip_atomic_store(&bar[1], g + 1u, __ATOMIC_RELEASE, __HIP_MEMORY_SCOPE_AGENT);
    } else {
      while (__hip_atomic_load(&bar[1], __ATOMIC_ACQUIRE, __HIP_MEMORY_SCOPE_AGENT) == g)
        __builtin_amdgcn_s_sleep(1);
    }
  }
  __syncthreads();
  __threadfence();
}

// One LSTM layer phase. Each wave owns one (mtile, jtile) 16x16 hidden tile and
// computes all four gate tiles (i,f,g,o) so the c/h update stays in registers.
// A panels are staged through LDS (all 8 waves of a block share the M-tile).
__device__ __forceinline__ void lstm_layer(
    int wave, int lane, unsigned short* Ash,
    const unsigned short* __restrict__ Ain, int ldA, int kItA,
    const unsigned short* Wih,                     // [2048][ldA]
    const unsigned short* __restrict__ Hprev,      // [128][512]
    const unsigned short* Whh,                     // [2048][512]
    const float* biasl,                            // [2048] (b_ih+b_hh)
    float* __restrict__ c,                         // [128][512]
    unsigned short* __restrict__ Hcur)             // [128][512]
{
  const int mt = wave >> 5, jt = wave & 31;
  const int cidx  = lane & 15;
  const int khalf = (lane >> 4) << 3;
  const int j = jt * 16 + cidx;

  f32x8 acc[4];
#pragma unroll
  for (int g = 0; g < 4; ++g) {
    float bv = biasl[g * H_ + j];                  // bias depends only on column
    acc[g] = (f32x8){bv, bv, bv, bv, bv, bv, bv, bv};
  }

  // ---- input GEMM: acc += A[m,:] * Wih[n,:]^T   (A via LDS) ----
  stage16(Ash, Ain + (size_t)mt * 16 * ldA, 16 * ldA);
  __syncthreads();
#pragma clang loop unroll_count(2)
  for (int kk = 0; kk < kItA; ++kk) {
    const int k0 = kk * 32 + khalf;
    bf16x16 a = load_frag(Ash, ldA, cidx, k0);     // ds_load_b128 x2
    if (kk + 1 < kItA)
      __builtin_prefetch(Wih + (size_t)j * ldA + (kk + 1) * 32, 0, 1);
#pragma unroll
    for (int g = 0; g < 4; ++g) {
      bf16x16 b = load_frag(Wih, ldA, g * H_ + j, k0);
      acc[g] = wmma_bf16(a, b, acc[g]);
    }
  }
  __syncthreads();

  // ---- recurrent GEMM: acc += Hprev[m,:] * Whh[n,:]^T   (K = 512) ----
  stage16(Ash, Hprev + (size_t)mt * 16 * H_, 16 * H_);
  __syncthreads();
#pragma clang loop unroll_count(2)
  for (int kk = 0; kk < H_ / 32; ++kk) {
    const int k0 = kk * 32 + khalf;
    bf16x16 a = load_frag(Ash, H_, cidx, k0);
#pragma unroll
    for (int g = 0; g < 4; ++g) {
      bf16x16 b = load_frag(Whh, H_, g * H_ + j, k0);
      acc[g] = wmma_bf16(a, b, acc[g]);
    }
  }

  // element-wise LSTM cell update; C-frag element r -> row (mt*16 + khalf + r)
  const int rbase = mt * 16 + khalf;
#pragma unroll
  for (int r = 0; r < 8; ++r) {
    const int idx = (rbase + r) * H_ + j;
    float cn = sigm(acc[1][r]) * c[idx] + sigm(acc[0][r]) * tanhf(acc[2][r]);
    c[idx] = cn;
    Hcur[idx] = f32_to_bf16u(sigm(acc[3][r]) * tanhf(cn));
  }
}

__global__ __launch_bounds__(TPB_) void rnn_persistent(
    const unsigned short* __restrict__ wih0p,
    const unsigned short* __restrict__ whh0,
    const unsigned short* __restrict__ wihr,
    const unsigned short* __restrict__ whhr,
    const unsigned short* __restrict__ decp,
    const unsigned short* __restrict__ encp,
    const float* __restrict__ biasc,
    const float* __restrict__ dec_b,
    const float* __restrict__ enc_b,
    unsigned short* xcat,
    unsigned short* hbuf,
    float* cbuf,
    unsigned short* logb,
    float* out,
    unsigned* bar)
{
  __shared__ unsigned short Ash[16 * K0P];         // 17,408 B staging panel

  const int tid   = blockIdx.x * blockDim.x + threadIdx.x;
  const int wave  = tid >> 5;
  const int lane  = threadIdx.x & 31;
  const int cidx  = lane & 15;
  const int khalf = (lane >> 4) << 3;

  for (int t = 0; t < T_; ++t) {
    // Fresh opaque zero per iteration: addresses "base + z_t" cannot be proven
    // invariant (no hoist/spill across t) but keep global address space.
    size_t z = 0; OPQ(z);
    const unsigned short* wih0_t = wih0p + z;
    const unsigned short* whh0_t = whh0 + z;
    const unsigned short* wih1_t = wihr + z;
    const unsigned short* whh1_t = whhr + z;
    const unsigned short* wih2_t = wihr + (size_t)G_ * H_ + z;
    const unsigned short* whh2_t = whhr + (size_t)G_ * H_ + z;
    const unsigned short* decp_t = decp + z;
    const unsigned short* encp_t = encp + z;
    const float*          bias_t = biasc + z;

    const int cur = t & 1, prv = cur ^ 1;
    unsigned short* h0c = hbuf + (size_t)(0 * 2 + cur) * (B_ * H_);
    unsigned short* h0p = hbuf + (size_t)(0 * 2 + prv) * (B_ * H_);
    unsigned short* h1c = hbuf + (size_t)(1 * 2 + cur) * (B_ * H_);
    unsigned short* h1p = hbuf + (size_t)(1 * 2 + prv) * (B_ * H_);
    unsigned short* h2c = hbuf + (size_t)(2 * 2 + cur) * (B_ * H_);
    unsigned short* h2p = hbuf + (size_t)(2 * 2 + prv) * (B_ * H_);

    lstm_layer(wave, lane, Ash, xcat, K0P, K0P / 32, wih0_t, h0p, whh0_t,
               bias_t, cbuf, h0c);
    grid_sync(bar, NBLK);
    lstm_layer(wave, lane, Ash, h0c, H_, H_ / 32, wih1_t, h1p, whh1_t,
               bias_t + G_, cbuf + B_ * H_, h1c);
    grid_sync(bar, NBLK);
    lstm_layer(wave, lane, Ash, h1c, H_, H_ / 32, wih2_t, h2p, whh2_t,
               bias_t + 2 * G_, cbuf + 2 * B_ * H_, h2c);
    grid_sync(bar, NBLK);

    // ---- decoder: logits[128,47] = h2 * dec_w^T + dec_b (24 tiles) ----
    if (wave < 24) {
      const int mt = wave / 3, nt = wave % 3;
      const int m = mt * 16 + cidx;
      const int n = nt * 16 + cidx;
      float bv = (n < O_) ? dec_b[n] : 0.0f;
      f32x8 acc = (f32x8){bv, bv, bv, bv, bv, bv, bv, bv};
#pragma clang loop unroll_count(2)
      for (int kk = 0; kk < H_ / 32; ++kk) {
        const int k0 = kk * 32 + khalf;
        bf16x16 a = load_frag(h2c, H_, m, k0);
        bf16x16 b = load_frag(decp_t, H_, n, k0);
        acc = wmma_bf16(a, b, acc);
      }
      const int rbase = mt * 16 + khalf;
#pragma unroll
      for (int r = 0; r < 8; ++r) {
        const int rr = rbase + r;
        logb[rr * KE + n] = f32_to_bf16u(acc[r]);
        if (n < O_) out[((size_t)rr * T_ + t) * O_ + n] = acc[r];
      }
    }
    grid_sync(bar, NBLK);

    // ---- encoder feedback: xcat[:, :512] = logits * enc_w^T + enc_b ----
    {
      const int mt = wave >> 5, nt = wave & 31;
      const int n = nt * 16 + cidx;
      float bv = enc_b[n];
      f32x8 acc = (f32x8){bv, bv, bv, bv, bv, bv, bv, bv};
      stage16(Ash, logb + (size_t)mt * 16 * KE, 16 * KE);
      __syncthreads();
#pragma unroll
      for (int kk = 0; kk < KE / 32; ++kk) {
        const int k0 = kk * 32 + khalf;
        bf16x16 a = load_frag(Ash, KE, cidx, k0);
        bf16x16 b = load_frag(encp_t, KE, n, k0);
        acc = wmma_bf16(a, b, acc);
      }
      const int rbase = mt * 16 + khalf;
#pragma unroll
      for (int r = 0; r < 8; ++r)
        xcat[(rbase + r) * K0P + n] = f32_to_bf16u(acc[r]);
    }
    grid_sync(bar, NBLK);
  }
}

// ---------------- prep kernels (run every launch; graph-safe) ----------------
__global__ void k_convert(const float* __restrict__ s, unsigned short* __restrict__ d, int n) {
  int i = blockIdx.x * blockDim.x + threadIdx.x;
  if (i < n) d[i] = f32_to_bf16u(s[i]);
}
__global__ void k_convert_pad(const float* __restrict__ s, unsigned short* __restrict__ d,
                              int dR, int sR, int dK, int sK) {
  int i = blockIdx.x * blockDim.x + threadIdx.x;
  if (i >= dR * dK) return;
  int r = i / dK, k = i % dK;
  d[i] = (r < sR && k < sK) ? f32_to_bf16u(s[r * sK + k]) : (unsigned short)0;
}
__global__ void k_zero16(unsigned short* p, int n) {
  int i = blockIdx.x * blockDim.x + threadIdx.x;
  if (i < n) p[i] = 0;
}
__global__ void k_zero32(float* p, int n) {
  int i = blockIdx.x * blockDim.x + threadIdx.x;
  if (i < n) p[i] = 0.0f;
}
__global__ void k_bias(const float* bi0, const float* bh0,
                       const float* bir, const float* bhr, float* bias) {
  int i = blockIdx.x * blockDim.x + threadIdx.x;
  if (i >= 3 * G_) return;
  int l = i / G_, g = i % G_;
  bias[i] = (l == 0) ? (bi0[g] + bh0[g])
                     : (bir[(l - 1) * G_ + g] + bhr[(l - 1) * G_ + g]);
}
// xcat init: encoded start token (one-hot at symbol 1) + properties + zero pad.
__global__ void k_init_xcat(const float* __restrict__ enc_w, const float* __restrict__ enc_b,
                            const float* __restrict__ props, unsigned short* xcat,
                            unsigned* bar) {
  int i = blockIdx.x * blockDim.x + threadIdx.x;
  if (i == 0) { bar[0] = 0u; bar[1] = 0u; }
  if (i >= B_ * K0P) return;
  int b = i / K0P, k = i % K0P;
  float v;
  if (k < H_)            v = enc_w[k * O_ + 1] + enc_b[k];   // onehot(1) @ enc_w^T
  else if (k < H_ + P_)  v = props[b * P_ + (k - H_)];
  else                   v = 0.0f;
  xcat[i] = f32_to_bf16u(v);
}

static inline unsigned cdiv(unsigned a, unsigned b) { return (a + b - 1) / b; }

extern "C" void kernel_launch(void* const* d_in, const int* in_sizes, int n_in,
                              void* d_out, int out_size, void* d_ws, size_t ws_size,
                              hipStream_t stream) {
  (void)in_sizes; (void)n_in; (void)out_size; (void)ws_size;
  const float* props = (const float*)d_in[1];
  const float* enc_w = (const float*)d_in[2];
  const float* enc_b = (const float*)d_in[3];
  const float* dec_w = (const float*)d_in[4];
  const float* dec_b = (const float*)d_in[5];
  const float* w_ih0 = (const float*)d_in[6];
  const float* w_hh0 = (const float*)d_in[7];
  const float* b_ih0 = (const float*)d_in[8];
  const float* b_hh0 = (const float*)d_in[9];
  const float* w_ihr = (const float*)d_in[10];
  const float* w_hhr = (const float*)d_in[11];
  const float* b_ihr = (const float*)d_in[12];
  const float* b_hhr = (const float*)d_in[13];

  char* ws = (char*)d_ws;
  unsigned short* wih0p = (unsigned short*)(ws + WS_WIH0);
  unsigned short* whh0b = (unsigned short*)(ws + WS_WHH0);
  unsigned short* wihrb = (unsigned short*)(ws + WS_WIHR);
  unsigned short* whhrb = (unsigned short*)(ws + WS_WHHR);
  unsigned short* decp  = (unsigned short*)(ws + WS_DECP);
  unsigned short* encp  = (unsigned short*)(ws + WS_ENCP);
  float*          biasc = (float*)(ws + WS_BIAS);
  unsigned short* xcat  = (unsigned short*)(ws + WS_XCAT);
  unsigned short* hbuf  = (unsigned short*)(ws + WS_HBUF);
  float*          cbuf  = (float*)(ws + WS_CBUF);
  unsigned short* logb  = (unsigned short*)(ws + WS_LOGB);
  unsigned*       bar   = (unsigned*)(ws + WS_BAR);

  const int TPB = TPB_;
  k_convert_pad<<<cdiv(G_ * K0P, TPB), TPB, 0, stream>>>(w_ih0, wih0p, G_, G_, K0P, H_ + P_);
  k_convert    <<<cdiv(G_ * H_, TPB), TPB, 0, stream>>>(w_hh0, whh0b, G_ * H_);
  k_convert    <<<cdiv(2 * G_ * H_, TPB), TPB, 0, stream>>>(w_ihr, wihrb, 2 * G_ * H_);
  k_convert    <<<cdiv(2 * G_ * H_, TPB), TPB, 0, stream>>>(w_hhr, whhrb, 2 * G_ * H_);
  k_convert_pad<<<cdiv(OPAD * H_, TPB), TPB, 0, stream>>>(dec_w, decp, OPAD, O_, H_, H_);
  k_convert_pad<<<cdiv(H_ * KE, TPB), TPB, 0, stream>>>(enc_w, encp, H_, H_, KE, O_);
  k_bias       <<<cdiv(3 * G_, TPB), TPB, 0, stream>>>(b_ih0, b_hh0, b_ihr, b_hhr, biasc);
  k_zero16     <<<cdiv(3 * 2 * B_ * H_, TPB), TPB, 0, stream>>>(hbuf, 3 * 2 * B_ * H_);
  k_zero32     <<<cdiv(3 * B_ * H_, TPB), TPB, 0, stream>>>(cbuf, 3 * B_ * H_);
  k_zero16     <<<cdiv(B_ * KE, TPB), TPB, 0, stream>>>(logb, B_ * KE);
  k_init_xcat  <<<cdiv(B_ * K0P, TPB), TPB, 0, stream>>>(enc_w, enc_b, props, xcat, bar);

  rnn_persistent<<<NBLK, TPB, 0, stream>>>(wih0p, whh0b, wihrb, whhrb, decp, encp,
                                           biasc, dec_b, enc_b, xcat, hbuf, cbuf,
                                           logb, (float*)d_out, bar);
}
